// DyGrEncoder_32847909880001
// MI455X (gfx1250) — matrix operand order repaired and checked by reading, not compiled, not run on hardware
//
#include <hip/hip_runtime.h>
#include <hip/hip_bf16.h>

// Problem constants (from reference): N,F,C,L,T,E
#define NN 50000
#define FF 128
#define CC 128
#define LLAYERS 2
#define TT 8
#define EE 400000

typedef __attribute__((ext_vector_type(16))) __bf16 v16bf;
typedef __attribute__((ext_vector_type(8)))  float  v8f;

__device__ __forceinline__ float sigmoidf_(float x) {
    return 1.0f / (1.0f + __expf(-x));
}

// ---------------------------------------------------------------------------
// Pre-pack an fp32 weight matrix into bf16 WMMA B-fragment layout.
// Bp layout: [Mtiles][4 k-chunks][32 lanes][16 halfs]
//   element (ct,kc,lane,j): out-col m = ct*16 + lane%16
//                           k = kc*32 + (j/8)*16 + (lane/16)*8 + j%8  (ISA B layout)
// wIsMK=1: W stored [M,K] row-major (B[k][m] = W[m*K+k], e.g. gru_Wih)
// wIsMK=0: W stored [K,M] row-major (B[k][m] = W[k*M+m], e.g. conv_W[i])
// ---------------------------------------------------------------------------
__global__ void prepack_w(const float* __restrict__ W, __bf16* __restrict__ Bp,
                          int M, int wIsMK) {
    int tid = blockIdx.x * blockDim.x + threadIdx.x;
    if (tid >= M * 128) return;
    int j    = tid & 15;
    int lane = (tid >> 4) & 31;
    int kc   = (tid >> 9) & 3;
    int ct   = tid >> 11;
    int m    = ct * 16 + (lane & 15);
    int k    = kc * 32 + (j >> 3) * 16 + (lane >> 4) * 8 + (j & 7);
    float v  = wIsMK ? W[(size_t)m * 128 + k] : W[(size_t)k * M + m];
    Bp[tid]  = (__bf16)v;
}

__device__ __forceinline__ void load_btile(v16bf (&b)[4], const __bf16* __restrict__ Bp,
                                           int ct, int lane) {
    const __bf16* p = Bp + ((size_t)(ct * 4) * 32 + lane) * 16;
    #pragma unroll
    for (int kc = 0; kc < 4; ++kc)
        b[kc] = *(const v16bf*)(p + kc * 512);   // 1024B stride between k-chunks
}

// ---------------------------------------------------------------------------
// WMMA GEMM:  Cmat[Nrows,M] (+)= A[Nrows,128] x Bpacked   (+ bias[M])
// Block = 256 threads = 8 waves; each wave owns one 16-row tile and sweeps
// all M/16 column tiles, K=128 as 4 chained v_wmma_f32_16x16x32_bf16.
// M is compile-time -> immediate store/load offsets.  Two tiles are
// processed per iteration with statically-named ping/pong B buffers so the
// next tile's loads overlap the current tile's WMMAs WITHOUT any dynamic
// array indexing (which would spill fragments to scratch).
// ---------------------------------------------------------------------------
template <int M, bool ACCUM, bool BIAS>
__global__ __launch_bounds__(256) void gemm_wmma(
    const float* __restrict__ A, const __bf16* __restrict__ Bp,
    float* __restrict__ Cmat, const float* __restrict__ bias, int Nrows)
{
    const int lane  = threadIdx.x & 31;
    const int wave  = threadIdx.x >> 5;
    const int tile  = blockIdx.x * 8 + wave;
    const int row0  = tile * 16;
    if (row0 >= Nrows) return;                 // wave-uniform: EXEC stays full
    const int laneLo = lane & 15;
    const int laneHi = lane >> 4;
    constexpr int Mtiles = M >> 4;
    static_assert((Mtiles & 1) == 0, "Mtiles must be even for ping/pong");

    // Build 4 A fragments (16x32 bf16 each) from fp32 row (ISA A layout:
    // lane-half holds K 0-7 and 16-23 of its 32-chunk).
    v16bf afrag[4];
    const float* arow = A + (size_t)(row0 + laneLo) * 128 + laneHi * 8;
    #pragma unroll
    for (int kc = 0; kc < 4; ++kc) {
        const float* p = arow + kc * 32;
        v16bf a;
        #pragma unroll
        for (int i = 0; i < 8; ++i) {
            a[i]     = (__bf16)p[i];
            a[8 + i] = (__bf16)p[16 + i];
        }
        afrag[kc] = a;
    }

    // Base pointer for this wave's C sub-block (row = row0 + laneHi*8 + r)
    float* crow = Cmat + (size_t)(row0 + laneHi * 8) * M + laneLo;

    v16bf bA[4], bB[4];
    load_btile(bA, Bp, 0, lane);

    #pragma unroll 1
    for (int ct = 0; ct < Mtiles; ct += 2) {
        // --- tile ct (uses bA); prefetch ct+1 into bB ---
        load_btile(bB, Bp, ct + 1, lane);
        {
            v8f acc = {0.f, 0.f, 0.f, 0.f, 0.f, 0.f, 0.f, 0.f};
            float* cb = crow + ct * 16;
            if (ACCUM) {
                #pragma unroll
                for (int r = 0; r < 8; ++r) acc[r] = cb[r * M];
            }
            #pragma unroll
            for (int kc = 0; kc < 4; ++kc)
                acc = __builtin_amdgcn_wmma_f32_16x16x32_bf16(
                    false, afrag[kc], false, bA[kc], (short)0, acc, false, false);
            if (BIAS) {
                float bv = bias[ct * 16 + laneLo];
                #pragma unroll
                for (int r = 0; r < 8; ++r) cb[r * M] = acc[r] + bv;
            } else {
                #pragma unroll
                for (int r = 0; r < 8; ++r) cb[r * M] = acc[r];
            }
        }
        // --- tile ct+1 (uses bB); prefetch ct+2 into bA ---
        if (ct + 2 < Mtiles)
            load_btile(bA, Bp, ct + 2, lane);
        {
            v8f acc = {0.f, 0.f, 0.f, 0.f, 0.f, 0.f, 0.f, 0.f};
            float* cb = crow + (ct + 1) * 16;
            if (ACCUM) {
                #pragma unroll
                for (int r = 0; r < 8; ++r) acc[r] = cb[r * M];
            }
            #pragma unroll
            for (int kc = 0; kc < 4; ++kc)
                acc = __builtin_amdgcn_wmma_f32_16x16x32_bf16(
                    false, afrag[kc], false, bB[kc], (short)0, acc, false, false);
            if (BIAS) {
                float bv = bias[(ct + 1) * 16 + laneLo];
                #pragma unroll
                for (int r = 0; r < 8; ++r) cb[r * M] = acc[r] + bv;
            } else {
                #pragma unroll
                for (int r = 0; r < 8; ++r) cb[r * M] = acc[r];
            }
        }
    }
}

// ---------------------------------------------------------------------------
// Scatter-add (segment_sum over undirected edges). One wave per directed
// edge: contiguous 512B gather of m[src], fp32 atomics into agg[dst].
// Both arrays are L2-resident (25.6MB each in 192MB L2).
// ---------------------------------------------------------------------------
__global__ void scatter_add(const float* __restrict__ m,
                            const int* __restrict__ eit,   // [2,E] for step t
                            float* __restrict__ agg, int E)
{
    long long tid = (long long)blockIdx.x * blockDim.x + threadIdx.x;
    int e2 = (int)(tid >> 5);
    int c4 = (int)(tid & 31) << 2;
    if (e2 >= 2 * E) return;
    int s, d;
    if (e2 < E) { s = eit[e2];         d = eit[E + e2]; }
    else        { s = eit[E + e2 - E]; d = eit[e2 - E]; }
    const float4 v = *(const float4*)(m + (size_t)s * CC + c4);
    float* dst = agg + (size_t)d * CC + c4;
    __hip_atomic_fetch_add(dst + 0, v.x, __ATOMIC_RELAXED, __HIP_MEMORY_SCOPE_AGENT);
    __hip_atomic_fetch_add(dst + 1, v.y, __ATOMIC_RELAXED, __HIP_MEMORY_SCOPE_AGENT);
    __hip_atomic_fetch_add(dst + 2, v.z, __ATOMIC_RELAXED, __HIP_MEMORY_SCOPE_AGENT);
    __hip_atomic_fetch_add(dst + 3, v.w, __ATOMIC_RELAXED, __HIP_MEMORY_SCOPE_AGENT);
}

// ---------------------------------------------------------------------------
// GRU cell elementwise: x = (1-z)*tanh(i_n + r*h_n) + z*x
// ---------------------------------------------------------------------------
__global__ void gru_update(float* __restrict__ x, const float* __restrict__ gi,
                           const float* __restrict__ gh, int Nrows)
{
    long long tid = (long long)blockIdx.x * blockDim.x + threadIdx.x;
    int n = (int)(tid >> 7);
    int c = (int)(tid & 127);
    if (n >= Nrows) return;
    size_t b3 = (size_t)n * 384;
    float ir = gi[b3 + c], iz = gi[b3 + 128 + c], in_ = gi[b3 + 256 + c];
    float hr = gh[b3 + c], hz = gh[b3 + 128 + c], hn  = gh[b3 + 256 + c];
    float r  = sigmoidf_(ir + hr);
    float z  = sigmoidf_(iz + hz);
    float nn = tanhf(in_ + r * hn);
    size_t xi = (size_t)n * CC + c;
    x[xi] = (1.0f - z) * nn + z * x[xi];
}

// ---------------------------------------------------------------------------
// LSTM cell elementwise; writes Hn, Cn and out[n, t, c] (output is [N,T,C]).
// ---------------------------------------------------------------------------
__global__ void lstm_update(const float* __restrict__ gates,
                            float* __restrict__ H, float* __restrict__ Cc,
                            float* __restrict__ out, int Nrows, int t)
{
    long long tid = (long long)blockIdx.x * blockDim.x + threadIdx.x;
    int n = (int)(tid >> 7);
    int c = (int)(tid & 127);
    if (n >= Nrows) return;
    size_t b4 = (size_t)n * 512;
    float ig = gates[b4 + c];
    float fg = gates[b4 + 128 + c];
    float gg = gates[b4 + 256 + c];
    float og = gates[b4 + 384 + c];
    size_t xi = (size_t)n * CC + c;
    float cn = sigmoidf_(fg) * Cc[xi] + sigmoidf_(ig) * tanhf(gg);
    float hn = sigmoidf_(og) * tanhf(cn);
    Cc[xi] = cn;
    H[xi]  = hn;
    out[(size_t)n * TT * CC + (size_t)t * CC + c] = hn;
}

// ---------------------------------------------------------------------------
extern "C" void kernel_launch(void* const* d_in, const int* in_sizes, int n_in,
                              void* d_out, int out_size, void* d_ws, size_t ws_size,
                              hipStream_t stream) {
    const float* X     = (const float*)d_in[0];
    const int*   EI    = (const int*)  d_in[1];
    const float* Wpool = (const float*)d_in[2];
    const float* convW = (const float*)d_in[3];
    const float* gWih  = (const float*)d_in[4];
    const float* gWhh  = (const float*)d_in[5];
    const float* gbih  = (const float*)d_in[6];
    const float* gbhh  = (const float*)d_in[7];
    const float* lWih  = (const float*)d_in[8];
    const float* lWhh  = (const float*)d_in[9];
    const float* lbih  = (const float*)d_in[10];
    const float* lbhh  = (const float*)d_in[11];
    float* out = (float*)d_out;

    // Workspace layout (fp32 activations, then bf16 packed weights)
    float* x     = (float*)d_ws;                      // [N,128] node state
    float* m     = x     + (size_t)NN * CC;           // [N,128] projected
    float* agg   = m     + (size_t)NN * CC;           // [N,128] scatter accum
    float* gh    = agg   + (size_t)NN * CC;           // [N,384]
    float* gates = gh    + (size_t)NN * 3 * CC;       // [N,512]; also holds gi [N,384]
    float* H     = gates + (size_t)NN * 4 * CC;       // [N,128]
    float* Cc    = H     + (size_t)NN * CC;           // [N,128]
    __bf16* bp_pool = (__bf16*)(Cc + (size_t)NN * CC);
    __bf16* bp_c0   = bp_pool + 128 * 128;
    __bf16* bp_c1   = bp_c0   + 128 * 128;
    __bf16* bp_gih  = bp_c1   + 128 * 128;
    __bf16* bp_ghh  = bp_gih  + 384 * 128;
    __bf16* bp_lih  = bp_ghh  + 384 * 128;
    __bf16* bp_lhh  = bp_lih  + 512 * 128;

    // Pack all weights to bf16 fragment layout (tiny; once per launch)
    prepack_w<<<64,  256, 0, stream>>>(Wpool,             bp_pool, 128, 1);
    prepack_w<<<64,  256, 0, stream>>>(convW,             bp_c0,   128, 0);
    prepack_w<<<64,  256, 0, stream>>>(convW + 128 * 128, bp_c1,   128, 0);
    prepack_w<<<192, 256, 0, stream>>>(gWih,              bp_gih,  384, 1);
    prepack_w<<<192, 256, 0, stream>>>(gWhh,              bp_ghh,  384, 1);
    prepack_w<<<256, 256, 0, stream>>>(lWih,              bp_lih,  512, 1);
    prepack_w<<<256, 256, 0, stream>>>(lWhh,              bp_lhh,  512, 1);

    hipMemsetAsync(H,  0, (size_t)NN * CC * sizeof(float), stream);
    hipMemsetAsync(Cc, 0, (size_t)NN * CC * sizeof(float), stream);

    const int GEMM_BLOCKS = (NN / 16 + 7) / 8;                    // 391
    const int EW_BLOCKS   = (NN * CC + 255) / 256;                // 25000
    const int SC_BLOCKS   = (int)(((long long)2 * EE * 32 + 255) / 256); // 100000

    for (int t = 0; t < TT; ++t) {
        const float* Xt  = X  + (size_t)t * NN * FF;
        const int*   eit = EI + (size_t)t * 2 * EE;

        // pool: x = Xt @ W_pool.T
        gemm_wmma<128, false, false><<<GEMM_BLOCKS, 256, 0, stream>>>(
            Xt, bp_pool, x, nullptr, NN);

        for (int l = 0; l < LLAYERS; ++l) {
            // m = x @ conv_W[l]
            gemm_wmma<128, false, false><<<GEMM_BLOCKS, 256, 0, stream>>>(
                x, (l == 0) ? bp_c0 : bp_c1, m, nullptr, NN);
            // agg = segment_sum(m[src], dst) over undirected edges
            hipMemsetAsync(agg, 0, (size_t)NN * CC * sizeof(float), stream);
            scatter_add<<<SC_BLOCKS, 256, 0, stream>>>(m, eit, agg, EE);
            // gi = agg @ Wih.T + bih  (stored in `gates` region, ld=384)
            gemm_wmma<384, false, true><<<GEMM_BLOCKS, 256, 0, stream>>>(
                agg, bp_gih, gates, gbih, NN);
            // gh = x @ Whh.T + bhh
            gemm_wmma<384, false, true><<<GEMM_BLOCKS, 256, 0, stream>>>(
                x, bp_ghh, gh, gbhh, NN);
            // GRU elementwise update of x
            gru_update<<<EW_BLOCKS, 256, 0, stream>>>(x, gates, gh, NN);
        }

        // LSTM gates = h_tilde @ Wih.T + bih + H @ Whh.T + bhh
        gemm_wmma<512, false, true><<<GEMM_BLOCKS, 256, 0, stream>>>(
            x, bp_lih, gates, lbih, NN);
        gemm_wmma<512, true, true><<<GEMM_BLOCKS, 256, 0, stream>>>(
            H, bp_lhh, gates, lbhh, NN);
        lstm_update<<<EW_BLOCKS, 256, 0, stream>>>(gates, H, Cc, out, NN, t);
    }
}
